// LSTMClassifier_28999619182911
// MI455X (gfx1250) — matrix operand order repaired and checked by reading, not compiled, not run on hardware
//
#include <hip/hip_runtime.h>
#include <stdint.h>

typedef __bf16 bf16_t;
typedef __attribute__((ext_vector_type(8)))  bf16_t v8bf;
typedef __attribute__((ext_vector_type(16))) bf16_t v16bf;
typedef __attribute__((ext_vector_type(8)))  float  v8f;

#define B_  128
#define T_  1024
#define D_  128
#define H_  256
#define G4_ 1024   // 4*H

// ---------- helpers ----------
__device__ __forceinline__ uint16_t f2bf(float f) {
  union { float f; uint32_t u; } x; x.f = f;
  uint32_t u = x.u + 0x7FFFu + ((x.u >> 16) & 1u);   // round-nearest-even
  return (uint16_t)(u >> 16);
}
__device__ __forceinline__ float sigmoid_f(float x) {
  return 1.0f / (1.0f + __expf(-x));
}
__device__ __forceinline__ float tanh_f(float x) {
  // 1 - 2/(e^{2x}+1): saturates correctly at +/-1 for large |x|
  float e = __expf(2.0f * x);
  return 1.0f - 2.0f / (e + 1.0f);
}

// ---------- pack [W;U] -> bf16, transposed to [G4 rows(n), Kc cols(k)] ----------
// wcatT[n*Kc + k] = (k < Din ? W[k,n] : U[k-Din,n]) so B-matrix fragments are
// contiguous 32B loads per lane (column-major U/W becomes row-contiguous in K).
__global__ void pack_wu(const float* __restrict__ W, const float* __restrict__ U,
                        uint16_t* __restrict__ out, int Din) {
  int Kc = Din + H_;
  int total = G4_ * Kc;
  int idx = blockIdx.x * blockDim.x + threadIdx.x;
  if (idx >= total) return;
  int n = idx / Kc, k = idx - n * Kc;
  float v = (k < Din) ? W[(size_t)k * G4_ + n] : U[(size_t)(k - Din) * G4_ + n];
  out[idx] = f2bf(v);
}

// ---------- fused LSTM layer scan ----------
// One block = 16 batch rows, entire T loop. 8 wave32s; wave w owns hidden
// columns j in [32w, 32w+32) across all 4 gates (8 WMMA column tiles), so
// gate recombination into (c, h) is wave-local in registers.
// A = [x_t | h_t] (bf16) staged in LDS; B streamed from L2-resident
// pre-transposed weights every step (rolled k-loop => no LICM hoist/spill).
template <int DIN, bool IN_F32, bool OUT_F32>
__global__ __launch_bounds__(256) void lstm_scan(
    const void* __restrict__ xin,        // [B,T,DIN] f32 or bf16
    const uint16_t* __restrict__ wcatT,  // [G4, KC] bf16, transposed
    const float* __restrict__ bias,      // [G4]
    void* __restrict__ hout)             // [B,T,H] bf16 or f32
{
  constexpr int KC  = DIN + H_;          // 384 or 512
  constexpr int KCP = KC + 8;            // +16B row pad for LDS bank spread
  __shared__ __align__(32) uint16_t As[16 * KCP];
  __shared__ int s_oz;                   // opaque zero (anti-LICM)

  const int tid   = threadIdx.x;
  const int wave  = tid >> 5;
  const int lane  = tid & 31;
  const int nlo   = lane & 15;           // N column within 16x16 tile
  const int khalf = lane >> 4;           // 0/1: which K half-chunk this lane holds
  const int b0    = blockIdx.x * 16;     // batch tile base

  if (tid == 0) s_oz = 0;
  // zero initial h region of LDS (visible after first barrier)
  for (int e = tid; e < 16 * H_; e += 256) {
    int m = e >> 8, j = e & (H_ - 1);
    As[m * KCP + DIN + j] = 0;
  }

  // bias per owned column tile (C-layout: lane's column = n0 + (lane&15))
  float bv[4][2];
#pragma unroll
  for (int g = 0; g < 4; ++g)
#pragma unroll
    for (int s = 0; s < 2; ++s)
      bv[g][s] = bias[g * H_ + wave * 32 + s * 16 + nlo];

  // cell state in registers: rows M = khalf*8 + r, col j = 32w + 16s + nlo
  v8f cst[2];
#pragma unroll
  for (int s = 0; s < 2; ++s)
#pragma unroll
    for (int r = 0; r < 8; ++r) cst[s][r] = 0.0f;

  const float*    xf  = (const float*)xin;
  const uint16_t* xb  = (const uint16_t*)xin;
  uint16_t*       hob = (uint16_t*)hout;
  float*          hof = (float*)hout;

  for (int t = 0; t < T_; ++t) {
    // ---- stage x_t into LDS A-buffer cols [0, DIN) ----
    for (int e = tid; e < 16 * DIN; e += 256) {
      int m = e / DIN, k = e - m * DIN;
      size_t src = ((size_t)(b0 + m) * T_ + t) * DIN + k;
      As[m * KCP + k] = IN_F32 ? f2bf(xf[src]) : xb[src];
    }
    __syncthreads();  // x_t and previous h_t visible to all waves

    // Opaque zero loaded from LDS *after* the barrier: the compiler cannot
    // forward the store across __syncthreads, so weight addresses are not
    // loop-invariant across t => no hoist-and-spill of B fragments.
    int oz = s_oz;
    // per-lane weight base: row (wave*32 + nlo), K half-chunk khalf*16
    const uint16_t* lanew =
        wcatT + oz + (size_t)(wave * 32 + nlo) * KC + khalf * 16;

    // ---- gates = [x_t | h] @ [W;U]^T-packed + b ----
    v8f acc[4][2];
#pragma unroll
    for (int g = 0; g < 4; ++g)
#pragma unroll
      for (int s = 0; s < 2; ++s)
#pragma unroll
        for (int r = 0; r < 8; ++r) acc[g][s][r] = bv[g][s];

#pragma unroll 1      // keep rolled: B loads must stay inside (no LICM/spill)
    for (int kb = 0; kb < KC; kb += 32) {
      // A fragment (16x32 bf16): lanes 0-15 rows M=lane hold K {0..15};
      // lanes 16-31 rows M=lane-16 hold K {16..31}
      int aoff = kb + khalf * 16;
      const uint16_t* arow = &As[nlo * KCP];
      v8bf alo = *(const v8bf*)(arow + aoff);
      v8bf ahi = *(const v8bf*)(arow + aoff + 8);
      v16bf av = __builtin_shufflevector(alo, ahi,
          0, 1, 2, 3, 4, 5, 6, 7, 8, 9, 10, 11, 12, 13, 14, 15);

      // batch all 8 B fragments first -> one load clause + one wait,
      // then 8 independent-accumulator WMMAs back-to-back
      v16bf bt[4][2];
#pragma unroll
      for (int g = 0; g < 4; ++g)
#pragma unroll
        for (int s = 0; s < 2; ++s)
          bt[g][s] = *(const v16bf*)(lanew +
                         (size_t)(g * H_ + s * 16) * KC + kb);

#pragma unroll
      for (int g = 0; g < 4; ++g)
#pragma unroll
        for (int s = 0; s < 2; ++s)
          acc[g][s] = __builtin_amdgcn_wmma_f32_16x16x32_bf16(
              false, av, false, bt[g][s], (short)0, acc[g][s], false, false);
    }

    // ---- nonlinearity + state update; waves own disjoint j-ranges ----
#pragma unroll
    for (int s = 0; s < 2; ++s) {
      int j = wave * 32 + s * 16 + nlo;
#pragma unroll
      for (int r = 0; r < 8; ++r) {
        float iv = sigmoid_f(acc[0][s][r]);
        float fv = sigmoid_f(acc[1][s][r]);
        float gv = tanh_f   (acc[2][s][r]);
        float ov = sigmoid_f(acc[3][s][r]);
        float c  = fv * cst[s][r] + iv * gv;
        cst[s][r] = c;
        float h  = ov * tanh_f(c);
        int m = khalf * 8 + r;
        As[m * KCP + DIN + j] = f2bf(h);               // feed next timestep
        size_t dst = ((size_t)(b0 + m) * T_ + t) * H_ + j;
        if (OUT_F32) hof[dst] = h;                     // layer 1 -> d_out
        else         hob[dst] = f2bf(h);               // layer 0 -> ws (bf16)
      }
    }
    __syncthreads();  // h writes complete before anyone stages t+1 / reads h
  }
}

// ---------- launch ----------
extern "C" void kernel_launch(void* const* d_in, const int* in_sizes, int n_in,
                              void* d_out, int out_size, void* d_ws, size_t ws_size,
                              hipStream_t stream) {
  const float* x  = (const float*)d_in[0];
  const float* W0 = (const float*)d_in[1];
  const float* U0 = (const float*)d_in[2];
  const float* b0 = (const float*)d_in[3];
  const float* W1 = (const float*)d_in[4];
  const float* U1 = (const float*)d_in[5];
  const float* b1 = (const float*)d_in[6];

  uint8_t* ws = (uint8_t*)d_ws;
  const size_t SZ_W0 = (size_t)G4_ * (D_ + H_) * sizeof(uint16_t);  // 0.75 MB
  const size_t SZ_W1 = (size_t)G4_ * (H_ + H_) * sizeof(uint16_t);  // 1 MB
  uint16_t* wcat0 = (uint16_t*)ws;
  uint16_t* wcat1 = (uint16_t*)(ws + SZ_W0);
  uint16_t* hseq0 = (uint16_t*)(ws + SZ_W0 + SZ_W1);  // [B,T,H] bf16, 64 MB

  int tot0 = G4_ * (D_ + H_);
  pack_wu<<<(tot0 + 255) / 256, 256, 0, stream>>>(W0, U0, wcat0, D_);
  int tot1 = G4_ * (H_ + H_);
  pack_wu<<<(tot1 + 255) / 256, 256, 0, stream>>>(W1, U1, wcat1, H_);

  // layer 0: fp32 x in, bf16 h-sequence out (feeds layer 1)
  lstm_scan<D_, true, false><<<B_ / 16, 256, 0, stream>>>(x, wcat0, b0, hseq0);
  // layer 1: bf16 h-sequence in, fp32 out -> d_out [B,T,H]
  lstm_scan<H_, false, true><<<B_ / 16, 256, 0, stream>>>(hseq0, wcat1, b1, d_out);
}
